// BioRNN_14113262535219
// MI455X (gfx1250) — matrix operand order repaired and checked by reading, not compiled, not run on hardware
//
#include <hip/hip_runtime.h>

typedef __attribute__((ext_vector_type(16))) __bf16 v16bf;
typedef __attribute__((ext_vector_type(8)))  float  v8f;

#define T_STEPS 1000
#define BATCH   4096
#define IN_SZ   8
#define HID     50
#define OUT_SZ  6
#define DT_C    0.1f

// d_ws layout (bytes)
#define WS_WCT  0       // [64][64] ushort bf16 : row j = [W_in(j,:8) | W_rec(j,:50) | 0]
#define WS_WOUT 8192    // [16][64] ushort bf16 : row n = [0(8) | W_out_w(n,:50) | 0]
#define WS_BIAS 10240   // [64] float (padded bias)
#define WS_BOUT 10496   // [16] float (padded W_out_b)

// RNE rounding: used only in the one-time weight prep
__device__ __forceinline__ unsigned short f32_bf16_rne(float f) {
    unsigned u = __float_as_uint(f);
    u += 0x7FFFu + ((u >> 16) & 1u);
    return (unsigned short)(u >> 16);
}

// Truncating f32->bf16 (zero ALU when fused into a d16_hi store): used on the
// serial critical path. The recurrence h <- 0.9h + 0.1(W r) is contractive, so
// the extra 2^-8 quantization noise does not accumulate.
__device__ __forceinline__ unsigned short f32_bf16_trunc(float f) {
    return (unsigned short)(__float_as_uint(f) >> 16);
}

__global__ void biornn_prep(const float* __restrict__ W_in,
                            const float* __restrict__ W_rec,
                            const float* __restrict__ bias,
                            const float* __restrict__ W_out_w,
                            const float* __restrict__ W_out_b,
                            void* __restrict__ ws)
{
    int i = blockIdx.x * blockDim.x + threadIdx.x;
    unsigned short* wct = (unsigned short*)((char*)ws + WS_WCT);
    unsigned short* wot = (unsigned short*)((char*)ws + WS_WOUT);
    float* bv = (float*)((char*)ws + WS_BIAS);
    float* bo = (float*)((char*)ws + WS_BOUT);

    if (i < 64 * 64) {                       // fused [W_in | W_rec] transposed-B
        int j = i >> 6, k = i & 63;
        float v = 0.f;
        if (j < HID) {
            if (k < IN_SZ)            v = W_in[j * IN_SZ + k];
            else if (k < IN_SZ + HID) v = W_rec[j * HID + (k - IN_SZ)];
        }
        wct[i] = f32_bf16_rne(v);
    } else if (i < 64 * 64 + 16 * 64) {      // padded W_out transposed-B
        int i2 = i - 64 * 64;
        int n = i2 >> 6, k = i2 & 63;
        float v = 0.f;
        if (n < OUT_SZ && k >= IN_SZ && k < IN_SZ + HID)
            v = W_out_w[n * HID + (k - IN_SZ)];
        wot[i2] = f32_bf16_rne(v);
    } else if (i < 64 * 64 + 16 * 64 + 64) {
        int j = i - (64 * 64 + 16 * 64);
        bv[j] = (j < HID) ? bias[j] : 0.f;
    } else if (i < 64 * 64 + 16 * 64 + 64 + 16) {
        int n = i - (64 * 64 + 16 * 64 + 64);
        bo[n] = (n < OUT_SZ) ? W_out_b[n] : 0.f;
    }
}

union FragU { uint4 q[2]; v16bf v; };

__global__ __launch_bounds__(32)
void biornn_main(const float* __restrict__ x,
                 const void* __restrict__ ws,
                 float* __restrict__ out)
{
    // 16 rows x 64 cols bf16, row stride 128B. Row m = [x(k=0..7) | r(k=8..57) | 0]
    __shared__ unsigned short ldsA[16 * 64];

    const int lane = threadIdx.x;
    const int lo = lane & 15, hi = lane >> 4;
    const int bbase = blockIdx.x * 16;

    const unsigned short* wct = (const unsigned short*)((const char*)ws + WS_WCT);
    const unsigned short* wot = (const unsigned short*)((const char*)ws + WS_WOUT);
    const float* bv = (const float*)((const char*)ws + WS_BIAS);
    const float* bo = (const float*)((const char*)ws + WS_BOUT);

    // ---- Pin all weight B-fragments in VGPRs for the whole T loop ----
    // B frag (16x16x32 bf16): lane<16 holds col n=lo, K=32c..32c+15; lane>=16: K=32c+16..+31
    FragU wc[4][2];
#pragma unroll
    for (int nt = 0; nt < 4; ++nt)
#pragma unroll
        for (int c = 0; c < 2; ++c) {
            const unsigned short* p = wct + (16 * nt + lo) * 64 + c * 32 + hi * 16;
            wc[nt][c].q[0] = *(const uint4*)(p);
            wc[nt][c].q[1] = *(const uint4*)(p + 8);
        }
    FragU wo[2];
#pragma unroll
    for (int c = 0; c < 2; ++c) {
        const unsigned short* p = wot + lo * 64 + c * 32 + hi * 16;
        wo[c].q[0] = *(const uint4*)(p);
        wo[c].q[1] = *(const uint4*)(p + 8);
    }

    float biasdt[4];
#pragma unroll
    for (int nt = 0; nt < 4; ++nt) biasdt[nt] = DT_C * bv[16 * nt + lo];
    const float bout = bo[lo];

    // zero the k=58..63 pad once (also refreshed every step by predicated r stores)
    if (hi == 0) {
        unsigned short* zp = &ldsA[lo * 64 + 58];
#pragma unroll
        for (int i = 0; i < 6; ++i) zp[i] = 0;
    }

    // hidden state: 4 C-tiles (16 batch rows x 64 hidden), f32
    v8f h[4];
#pragma unroll
    for (int nt = 0; nt < 4; ++nt) h[nt] = {};

    // x staging: 32 lanes cover 16 rows x 32B; lane -> (row, half)
    const int xrow = lane >> 1, xhalf = lane & 1;

#pragma unroll 1
    for (int t = 0; t < T_STEPS; ++t) {
        // 1. load x_t tile (coalesced 512B), truncate-pack f32->bf16 with v_perm,
        //    store into LDS k=0..7
        const float* xp = x + (((size_t)t * BATCH) + bbase + xrow) * IN_SZ + xhalf * 4;
        float4 xv = *(const float4*)xp;
        __builtin_prefetch(xp + (size_t)BATCH * IN_SZ, 0, 0);  // next step's x
        uint2 pk;
        // [hi16(y) | hi16(x)] in one v_perm_b32: S0=y, S1=x, sel 0x07060302
        pk.x = __builtin_amdgcn_perm(__float_as_uint(xv.y), __float_as_uint(xv.x),
                                     0x07060302u);
        pk.y = __builtin_amdgcn_perm(__float_as_uint(xv.w), __float_as_uint(xv.z),
                                     0x07060302u);
        *(uint2*)&ldsA[xrow * 64 + xhalf * 4] = pk;

        // 2. r = relu(h) -> bf16(trunc) -> LDS k=8..63 (C-layout -> row-major relayout)
        //    v_max_f32 + ds_store_b16_d16_hi per element; no conversion ALU.
        {
            unsigned short* rp = &ldsA[hi * 512 + 8 + lo];   // (m=8*hi, k=8+lo)
#pragma unroll
            for (int nt = 0; nt < 4; ++nt) {
                if (nt < 3 || lo < 8) {                      // keep k <= 63
#pragma unroll
                    for (int v = 0; v < 8; ++v) {
                        float hv = h[nt][v];
                        hv = hv > 0.f ? hv : 0.f;
                        rp[v * 64 + nt * 16] = f32_bf16_trunc(hv);
                    }
                }
            }
        }

        // 3. A fragments (DS in-order within wave; 4x ds_load_b128)
        // A lane<16: m=lo, K=32c+0..15; lane>=16: K=32c+16..31
        FragU a0, a1;
        {
            const unsigned short* ap = &ldsA[lo * 64];
            a0.q[0] = *(const uint4*)(ap + hi * 16);
            a0.q[1] = *(const uint4*)(ap + 8 + hi * 16);
            a1.q[0] = *(const uint4*)(ap + 32 + hi * 16);
            a1.q[1] = *(const uint4*)(ap + 32 + 8 + hi * 16);
        }

        // 4. WMMA: dh = [x|r] @ [W_in;W_rec]^T  (8 wmma), y = [x|r] @ Wout_pad^T (2 wmma)
        v8f acc[4];
#pragma unroll
        for (int nt = 0; nt < 4; ++nt) {
            v8f c = {};
            c = __builtin_amdgcn_wmma_f32_16x16x32_bf16(false, a0.v, false, wc[nt][0].v,
                                                        (short)0, c, false, false);
            c = __builtin_amdgcn_wmma_f32_16x16x32_bf16(false, a1.v, false, wc[nt][1].v,
                                                        (short)0, c, false, false);
            acc[nt] = c;
        }
        v8f yacc = {};
        yacc = __builtin_amdgcn_wmma_f32_16x16x32_bf16(false, a0.v, false, wo[0].v,
                                                       (short)0, yacc, false, false);
        yacc = __builtin_amdgcn_wmma_f32_16x16x32_bf16(false, a1.v, false, wo[1].v,
                                                       (short)0, yacc, false, false);

        // 5. y_t store: C-layout lane lo = output col (valid < 6), VGPR v -> row v+8*hi
        if (lo < OUT_SZ) {
            size_t ob = (((size_t)t * BATCH) + bbase + hi * 8) * OUT_SZ + lo;
#pragma unroll
            for (int v = 0; v < 8; ++v)
                out[ob + (size_t)v * OUT_SZ] = yacc[v] + bout;
        }

        // 6. h = (1-dt)*h + dt*(x@W_in^T + r@W_rec^T) + dt*bias
#pragma unroll
        for (int nt = 0; nt < 4; ++nt)
            h[nt] = h[nt] * (1.f - DT_C) + acc[nt] * DT_C + biasdt[nt];
    }
}

extern "C" void kernel_launch(void* const* d_in, const int* in_sizes, int n_in,
                              void* d_out, int out_size, void* d_ws, size_t ws_size,
                              hipStream_t stream) {
    const float* x       = (const float*)d_in[0];   // (1000, 4096, 8)
    const float* W_in    = (const float*)d_in[1];   // (50, 8)
    const float* W_rec   = (const float*)d_in[2];   // (50, 50)
    const float* bias    = (const float*)d_in[3];   // (50,)
    const float* W_out_w = (const float*)d_in[4];   // (6, 50)
    const float* W_out_b = (const float*)d_in[5];   // (6,)
    float* out = (float*)d_out;                     // (1000, 4096, 6)

    const int prep_elems = 64 * 64 + 16 * 64 + 64 + 16;
    biornn_prep<<<(prep_elems + 255) / 256, 256, 0, stream>>>(
        W_in, W_rec, bias, W_out_w, W_out_b, d_ws);

    // 4096 batch rows / 16 per wave = 256 single-wave workgroups, fully independent
    biornn_main<<<BATCH / 16, 32, 0, stream>>>(x, d_ws, out);
}
